// BinaryDenseLayer_25606595018967
// MI455X (gfx1250) — matrix operand order repaired and checked by reading, not compiled, not run on hardware
//
#include <hip/hip_runtime.h>

// ---------------------------------------------------------------------------
// BinaryDenseLayer: out = x @ sign(W) + b   (4096 x 4096 x 4096, f32 in/out)
//
// Fast path (needs 64MB workspace):
//   prep1: Xh  = f16(x)            [M][K] f16   (32 MB)  -> ws
//   prep2: WqT = f16(sign(W))^T    [N][K] f16   (32 MB)  -> ws + 32MB
//   gemm : f16 WMMA GEMM, 4 waves/block, 64x64 C tile per wave,
//          double-buffered LDS filled by GLOBAL_LOAD_ASYNC_TO_LDS_B128
//          (ASYNCcnt), V_WMMA_F32_16X16X32_F16, f32 accum.
// Fallback (small ws): fused convert+GEMM.
// ---------------------------------------------------------------------------

typedef _Float16 v4h  __attribute__((ext_vector_type(4)));
typedef _Float16 v8h  __attribute__((ext_vector_type(8)));
typedef _Float16 v16h __attribute__((ext_vector_type(16)));
typedef float    v8f  __attribute__((ext_vector_type(8)));

static __device__ __forceinline__ v16h cat8(v8h lo, v8h hi) {
  return __builtin_shufflevector(lo, hi, 0, 1, 2, 3, 4, 5, 6, 7,
                                         8, 9, 10, 11, 12, 13, 14, 15);
}

static __device__ __forceinline__ _Float16 sign_f16(float v) {
  return (v > 0.0f) ? (_Float16)1.0f
                    : ((v < 0.0f) ? (_Float16)-1.0f : (_Float16)0.0f);
}

// Async global->LDS copy of 16 bytes; same immediate offset applies to both
// the LDS destination and the global source (dsaddr = VDST + IOFFSET,
// gaddr = VADDR + IOFFSET). Tracked by ASYNCcnt.
static __device__ __forceinline__ void async_copy16(unsigned lds_addr,
                                                    const void* gaddr,
                                                    int ioff) {
  asm volatile("global_load_async_to_lds_b128 %0, %1, off offset:%2"
               :
               : "v"(lds_addr), "v"(gaddr), "i"(ioff)
               : "memory");
}

static __device__ __forceinline__ void wait_async0() {
  asm volatile("s_wait_asynccnt 0x0" ::: "memory");
}

#define MDIM 4096
#define NDIM 4096
#define KDIM 4096
#define BM 128
#define BN 128
#define BK 32
#define KTILES (KDIM / BK)

// ---------------------------------------------------------------------------
// Prep 1: x (f32, [M][K]) -> Xh (f16, [M][K]). Pure streaming convert.
// ---------------------------------------------------------------------------
__global__ __launch_bounds__(256)
void convert_x_kernel(const float* __restrict__ X, _Float16* __restrict__ Xh) {
  size_t i = ((size_t)blockIdx.x * 256 + threadIdx.x) * 4;
  float4 f = *(const float4*)(X + i);
  v4h h = {(_Float16)f.x, (_Float16)f.y, (_Float16)f.z, (_Float16)f.w};
  *(v4h*)(Xh + i) = h;
}

// ---------------------------------------------------------------------------
// Prep 2: W (f32, [K][N]) -> WqT (f16 sign, [N][K]).  64x64 LDS tile transpose.
// ---------------------------------------------------------------------------
#define TP 64
__global__ __launch_bounds__(256)
void convert_w_kernel(const float* __restrict__ W, _Float16* __restrict__ WqT) {
  __shared__ __align__(16) _Float16 tile[TP][TP + 8];  // padded rows (144B)

  const int tid = threadIdx.x;
  const int kb = blockIdx.x * TP;  // K block
  const int nb = blockIdx.y * TP;  // N block

  // Read phase: coalesced along N. 16 threads/row * 4 floats = 64 cols.
  const int rk = tid >> 4;         // 0..15 (+16*i)
  const int cn = (tid & 15) * 4;   // 0..60
#pragma unroll
  for (int i = 0; i < 4; ++i) {
    float4 w = *(const float4*)(W + (size_t)(kb + rk + 16 * i) * NDIM + nb + cn);
    v4h s = {sign_f16(w.x), sign_f16(w.y), sign_f16(w.z), sign_f16(w.w)};
    *(v4h*)&tile[rk + 16 * i][cn] = s;
  }
  __syncthreads();

  // Write phase: coalesced along K in the transposed output.
  const int rn = tid >> 4;         // output row n (0..15, +16*i)
  const int ck = (tid & 15) * 4;   // output col k
#pragma unroll
  for (int i = 0; i < 4; ++i) {
    int n = rn + 16 * i;
    v4h h = {tile[ck + 0][n], tile[ck + 1][n], tile[ck + 2][n], tile[ck + 3][n]};
    *(v4h*)(WqT + (size_t)(nb + n) * KDIM + kb + ck) = h;
  }
}

// ---------------------------------------------------------------------------
// Main GEMM: out[M][N] = Xh[M][K] * WqT[N][K]^T + bias.
// 128 threads = 4 waves; each wave owns a 64x64 C tile (4x4 WMMA accum).
// Tiles land in LDS via async copies; no VGPR staging in the hot loop.
// ---------------------------------------------------------------------------
__global__ __launch_bounds__(128)
void gemm_f16_wmma_kernel(const _Float16* __restrict__ Xh,
                          const _Float16* __restrict__ WqT,
                          const float* __restrict__ bias,
                          float* __restrict__ out) {
  __shared__ __align__(16) _Float16 As[2][BM][BK];  // [m][k]
  __shared__ __align__(16) _Float16 Bs[2][BN][BK];  // [n][k]

  const int tid  = threadIdx.x;
  const int lane = tid & 31;
  const int wid  = tid >> 5;  // 4 waves
  const int wm   = wid & 1;   // 2 strips of 64 C-rows
  const int wn   = wid >> 1;  // 2 strips of 64 C-cols

  const int row_blk = blockIdx.y * BM;
  const int col_blk = blockIdx.x * BN;

  // Copy split: one tile row (32 f16 = 64B = 4 x b128) per thread per tile.
  const _Float16* arow = Xh + (size_t)(row_blk + tid) * KDIM;
  const _Float16* brow = WqT + (size_t)(col_blk + tid) * KDIM;

  // LDS byte addresses of this thread's destination rows (addr[31:0] of the
  // generic address is the LDS offset per the aperture mapping).
  const unsigned lds_a0 = (unsigned)(uintptr_t)&As[0][tid][0];
  const unsigned lds_b0 = (unsigned)(uintptr_t)&Bs[0][tid][0];
  const unsigned buf_stride = (unsigned)((uintptr_t)&As[1][0][0] -
                                         (uintptr_t)&As[0][0][0]);

  v8f acc[4][4];
#pragma unroll
  for (int mi = 0; mi < 4; ++mi)
#pragma unroll
    for (int ni = 0; ni < 4; ++ni)
      acc[mi][ni] = 0;

  auto async_fill = [&](int buf, int k0) {
    const unsigned la = lds_a0 + (unsigned)buf * buf_stride;
    const unsigned lb = lds_b0 + (unsigned)buf * buf_stride;
    const _Float16* ga = arow + k0;
    const _Float16* gb = brow + k0;
#pragma unroll
    for (int j = 0; j < 4; ++j) {
      async_copy16(la, ga, 16 * j);   // advances LDS and global together
      async_copy16(lb, gb, 16 * j);
    }
  };

  auto compute = [&](int buf) {
    const _Float16* Ab = &As[buf][0][0];
    const _Float16* Bb = &Bs[buf][0][0];
    // A frag (16x32 f16): lanes 0-15 -> K 0..7 & 16..23,
    //                     lanes 16-31 -> K 8..15 & 24..31  (ISA 7.12.2)
    v16h af[4];
#pragma unroll
    for (int mi = 0; mi < 4; ++mi) {
      int row = wm * 64 + mi * 16 + (lane & 15);
      int k0  = (lane < 16) ? 0 : 8;
      af[mi] = cat8(*(const v8h*)(Ab + row * BK + k0),
                    *(const v8h*)(Ab + row * BK + k0 + 16));
    }
#pragma unroll
    for (int ni = 0; ni < 4; ++ni) {
      // B frag (32x16 f16 from [n][k] tile): lanes 0-15 -> K 0..7 & 16..23,
      //                                      lanes 16-31 -> K 8..15 & 24..31
      int row = wn * 64 + ni * 16 + (lane & 15);
      int k0  = (lane < 16) ? 0 : 8;
      v16h bf = cat8(*(const v8h*)(Bb + row * BK + k0),
                     *(const v8h*)(Bb + row * BK + k0 + 16));
#pragma unroll
      for (int mi = 0; mi < 4; ++mi) {
        acc[mi][ni] = __builtin_amdgcn_wmma_f32_16x16x32_f16(
            false, af[mi], false, bf, (short)0, acc[mi][ni], false, false);
      }
    }
  };

  // Prologue: fill buffer 0 and wait for it.
  async_fill(0, 0);
  wait_async0();
  __syncthreads();

  for (int kt = 0; kt < KTILES; ++kt) {
    const int buf = kt & 1;
    if (kt + 2 < KTILES) {  // L2 prefetch two tiles ahead (global_prefetch_b8)
      __builtin_prefetch(arow + (kt + 2) * BK, 0, 0);
      __builtin_prefetch(brow + (kt + 2) * BK, 0, 0);
    }
    if (kt + 1 < KTILES) async_fill(buf ^ 1, (kt + 1) * BK);
    compute(buf);                         // 16 WMMAs per wave
    if (kt + 1 < KTILES) wait_async0();   // our async writes landed in LDS
    __syncthreads();                      // everyone's writes + reads fenced
  }

  // Epilogue: VGPR r holds M=r (lanes 0-15) / M=8+r (lanes 16-31); N = lane&15.
#pragma unroll
  for (int mi = 0; mi < 4; ++mi) {
#pragma unroll
    for (int ni = 0; ni < 4; ++ni) {
      int m0 = row_blk + wm * 64 + mi * 16 + ((lane >> 4) << 3);
      int n0 = col_blk + wn * 64 + ni * 16 + (lane & 15);
      float bv = bias[n0];
      v8f c = acc[mi][ni];
#pragma unroll
      for (int r = 0; r < 8; ++r)
        out[(size_t)(m0 + r) * NDIM + n0] = c[r] + bv;
    }
  }
}

// ---------------------------------------------------------------------------
// Fallback: fused convert+GEMM (no workspace needed). 256 thr, 8 waves, 32x64.
// ---------------------------------------------------------------------------
__global__ __launch_bounds__(256)
void fused_gemm_kernel(const float* __restrict__ X, const float* __restrict__ W,
                       const float* __restrict__ bias, float* __restrict__ out) {
  __shared__ __align__(16) _Float16 As[2][BM][BK];
  __shared__ __align__(16) _Float16 Bs[2][BN][BK];

  const int tid  = threadIdx.x;
  const int lane = tid & 31;
  const int wid  = tid >> 5;
  const int wm   = wid & 3;
  const int wn   = wid >> 2;

  const int row_blk = blockIdx.y * BM;
  const int col_blk = blockIdx.x * BN;

  const int a_r = tid >> 3;
  const int a_c = (tid & 7) * 4;
  const int b_k = tid >> 5;
  const int b_n = (tid & 31) * 4;

  float4 a_reg[4];
  float4 b_reg[4];

  v8f acc[2][4];
#pragma unroll
  for (int mi = 0; mi < 2; ++mi)
#pragma unroll
    for (int ni = 0; ni < 4; ++ni)
      acc[mi][ni] = 0;

  auto gload = [&](int k0) {
#pragma unroll
    for (int i = 0; i < 4; ++i) {
      a_reg[i] = *(const float4*)(X + (size_t)(row_blk + a_r + 32 * i) * KDIM + k0 + a_c);
      b_reg[i] = *(const float4*)(W + (size_t)(k0 + b_k + 8 * i) * NDIM + col_blk + b_n);
    }
  };

  auto lstore = [&](int buf) {
#pragma unroll
    for (int i = 0; i < 4; ++i) {
      float4 a = a_reg[i];
      v4h ah = {(_Float16)a.x, (_Float16)a.y, (_Float16)a.z, (_Float16)a.w};
      *(v4h*)&As[buf][a_r + 32 * i][a_c] = ah;
      float4 w = b_reg[i];
      float wf[4] = {w.x, w.y, w.z, w.w};
#pragma unroll
      for (int j = 0; j < 4; ++j)
        Bs[buf][b_n + j][b_k + 8 * i] = sign_f16(wf[j]);
    }
  };

  auto compute = [&](int buf) {
    const _Float16* Ab = &As[buf][0][0];
    const _Float16* Bb = &Bs[buf][0][0];
    v16h af[2];
#pragma unroll
    for (int mi = 0; mi < 2; ++mi) {
      int row = wm * 32 + mi * 16 + (lane & 15);
      int k0  = (lane < 16) ? 0 : 8;
      af[mi] = cat8(*(const v8h*)(Ab + row * BK + k0),
                    *(const v8h*)(Ab + row * BK + k0 + 16));
    }
#pragma unroll
    for (int ni = 0; ni < 4; ++ni) {
      int row = wn * 64 + ni * 16 + (lane & 15);
      int k0  = (lane < 16) ? 0 : 8;
      v16h bf = cat8(*(const v8h*)(Bb + row * BK + k0),
                     *(const v8h*)(Bb + row * BK + k0 + 16));
#pragma unroll
      for (int mi = 0; mi < 2; ++mi) {
        acc[mi][ni] = __builtin_amdgcn_wmma_f32_16x16x32_f16(
            false, af[mi], false, bf, (short)0, acc[mi][ni], false, false);
      }
    }
  };

  gload(0);
  lstore(0);
  __syncthreads();

  for (int kt = 0; kt < KTILES; ++kt) {
    const int buf = kt & 1;
    if (kt + 1 < KTILES) gload((kt + 1) * BK);
    compute(buf);
    if (kt + 1 < KTILES) lstore(buf ^ 1);
    __syncthreads();
  }

#pragma unroll
  for (int mi = 0; mi < 2; ++mi) {
#pragma unroll
    for (int ni = 0; ni < 4; ++ni) {
      int m0 = row_blk + wm * 32 + mi * 16 + ((lane >> 4) << 3);
      int n0 = col_blk + wn * 64 + ni * 16 + (lane & 15);
      float bv = bias[n0];
      v8f c = acc[mi][ni];
#pragma unroll
      for (int r = 0; r < 8; ++r)
        out[(size_t)(m0 + r) * NDIM + n0] = c[r] + bv;
    }
  }
}

// ---------------------------------------------------------------------------
extern "C" void kernel_launch(void* const* d_in, const int* in_sizes, int n_in,
                              void* d_out, int out_size, void* d_ws, size_t ws_size,
                              hipStream_t stream) {
  const float* X = (const float*)d_in[0];  // x  [4096, 4096] f32
  const float* W = (const float*)d_in[1];  // W  [4096, 4096] f32
  const float* b = (const float*)d_in[2];  // b  [4096] f32
  float* out = (float*)d_out;              // [4096, 4096] f32

  const size_t xh_bytes = (size_t)MDIM * KDIM * sizeof(_Float16);
  const size_t wt_bytes = (size_t)NDIM * KDIM * sizeof(_Float16);

  if (ws_size >= xh_bytes + wt_bytes) {
    _Float16* Xh  = (_Float16*)d_ws;
    _Float16* WqT = (_Float16*)((char*)d_ws + xh_bytes);

    // prep: one-time conversions (memory-bound, ~8us total at 23.3 TB/s)
    convert_x_kernel<<<dim3((MDIM * (size_t)KDIM) / (256 * 4)), dim3(256), 0,
                       stream>>>(X, Xh);
    convert_w_kernel<<<dim3(KDIM / TP, NDIM / TP), dim3(256), 0, stream>>>(W, WqT);

    // main f16 WMMA GEMM
    gemm_f16_wmma_kernel<<<dim3(NDIM / BN, MDIM / BM), dim3(128), 0, stream>>>(
        Xh, WqT, b, out);
  } else {
    fused_gemm_kernel<<<dim3(NDIM / BN, MDIM / BM), dim3(256), 0, stream>>>(
        X, W, b, out);
  }
}
